// ModulatedConv2d_28638841930307
// MI455X (gfx1250) — matrix operand order repaired and checked by reading, not compiled, not run on hardware
//
#include <hip/hip_runtime.h>

// ---------------- problem constants ----------------
#define B_N    16
#define C_IN   512
#define C_OUT  512
#define H_DIM  64
#define W_DIM  64
#define LATENT 512
// runtime_coef = sqrt(2)/sqrt(9*512) = sqrt(1/2304) = 1/48 exactly
#define W_COEF (1.0f / 48.0f)
#define FC_COEF 0.044194173824159216f   // 1/sqrt(512)
#define EPS_F 1e-8f

typedef __attribute__((ext_vector_type(16))) __bf16          v16bf;
typedef __attribute__((ext_vector_type(8)))  unsigned int    v8ui;
typedef __attribute__((ext_vector_type(8)))  float           v8f;

#define A_STRIDE 40   // ushorts per LDS row: 80B = 16B-aligned, 20-bank stride
#define B_STRIDE 40

__device__ __forceinline__ unsigned short f2bf(float f) {
  unsigned int u = __builtin_bit_cast(unsigned int, f);
  u += 0x7FFFu + ((u >> 16) & 1u);      // round-to-nearest-even
  return (unsigned short)(u >> 16);
}

// Build a 16xbf16 WMMA operand from two 16B LDS runs (2x ds_load_b128)
__device__ __forceinline__ v16bf frag_from(const unsigned short* lo,
                                           const unsigned short* hi) {
  uint4 a = *(const uint4*)lo;
  uint4 b = *(const uint4*)hi;
  v8ui v;
  v[0] = a.x; v[1] = a.y; v[2] = a.z; v[3] = a.w;
  v[4] = b.x; v[5] = b.y; v[6] = b.z; v[7] = b.w;
  return __builtin_bit_cast(v16bf, v);
}

// ---------------- prologue kernels ----------------

// s[b][c] = sum_l y[b][l] * fc_w[l][c] * FC_COEF + bias[c] + 1
__global__ __launch_bounds__(256)
void style_fc_kernel(const float* __restrict__ y, const float* __restrict__ fc_w,
                     const float* __restrict__ bias, float* __restrict__ s) {
  int tid = blockIdx.x * blockDim.x + threadIdx.x;     // B_N*C_IN threads
  int b = tid >> 9;
  int c = tid & (C_IN - 1);
  float acc = 0.0f;
  const float* yrow = y + b * LATENT;
  for (int l = 0; l < LATENT; ++l)
    acc += yrow[l] * fc_w[l * C_IN + c];
  s[tid] = acc * FC_COEF + bias[c] + 1.0f;
}

// wsq[c][o] = sum_{9 taps} (w*W_COEF)^2
__global__ __launch_bounds__(256)
void wsq_kernel(const float* __restrict__ w, float* __restrict__ wsq) {
  int tid = blockIdx.x * blockDim.x + threadIdx.x;     // C_IN*C_OUT threads
  int c = tid >> 9;
  int o = tid & (C_OUT - 1);
  float acc = 0.0f;
  for (int t = 0; t < 9; ++t) {
    float v = w[((size_t)t * C_IN + c) * C_OUT + o] * W_COEF;
    acc += v * v;
  }
  wsq[tid] = acc;
}

// d[b][o] = rsqrt(sum_c s[b][c]^2 * wsq[c][o] + eps)
__global__ __launch_bounds__(256)
void demod_kernel(const float* __restrict__ s, const float* __restrict__ wsq,
                  float* __restrict__ d) {
  int tid = blockIdx.x * blockDim.x + threadIdx.x;     // B_N*C_OUT threads
  int b = tid >> 9;
  int o = tid & (C_OUT - 1);
  const float* srow = s + b * C_IN;
  float acc = 0.0f;
  for (int c = 0; c < C_IN; ++c) {
    float sv = srow[c];
    acc += sv * sv * wsq[c * C_OUT + o];
  }
  d[tid] = rsqrtf(acc + EPS_F);
}

// One-time weight transpose+convert: wbf[t][o][c] = bf16(w[t][c][o] / 48)
// grid (9, 2), 256 threads. Reads lane-coalesced over o; 16B writes per thread.
__global__ __launch_bounds__(256)
void wprep_kernel(const float* __restrict__ w, unsigned short* __restrict__ wbf) {
  int t = blockIdx.x;                         // tap 0..8
  int o = blockIdx.y * 256 + threadIdx.x;     // cout 0..511
  unsigned short* dst = wbf + ((size_t)t * C_OUT + o) * C_IN;
  for (int c = 0; c < C_IN; c += 8) {
    union { unsigned short us[8]; uint4 v; } pk;
    #pragma unroll
    for (int j = 0; j < 8; ++j)
      pk.us[j] = f2bf(w[((size_t)t * C_IN + c + j) * C_OUT + o] * W_COEF);
    *(uint4*)(dst + c) = pk.v;
  }
}

// ---------------- main implicit-GEMM WMMA conv ----------------
// grid: (C_OUT/64, H, B); block: 64 threads = 2 wave32
// block computes out[b, o0:o0+64, h, 0:64]; each wave owns 32 pixels x 64 couts
__global__ __launch_bounds__(64)
void modconv_wmma_kernel(const float* __restrict__ x,
                         const unsigned short* __restrict__ wbf,
                         const float* __restrict__ s, const float* __restrict__ dvec,
                         float* __restrict__ out) {
  const int o0   = blockIdx.x * 64;
  const int h    = blockIdx.y;
  const int b    = blockIdx.z;
  const int tid  = threadIdx.x;
  const int lane = tid & 31;
  const int wv   = tid >> 5;
  const int m0   = wv * 32;               // this wave's 32 pixel columns

  // Fragment-major LDS staging (bf16):
  //   Ax[wcol][chan]  : lane A-slice = two contiguous 16B runs per pixel col
  //   Bw[kw][o][chan] : lane B-slice = two contiguous 16B runs per cout col
  __shared__ __align__(16) unsigned short Ax[66][A_STRIDE];
  __shared__ __align__(16) unsigned short Bw[3][64][B_STRIDE];

  v8f acc[2][4];
  #pragma unroll
  for (int i = 0; i < 2; ++i)
    #pragma unroll
    for (int j = 0; j < 4; ++j)
      acc[i][j] = (v8f){};

  // per-lane fragment constants (ISA 16-bit A/B lane layouts, 7.12.2)
  const int mlo    = lane & 15;
  const int khalfA = (lane & 16) ? 8 : 0;
  const int khalfB = (lane & 16) ? 8 : 0;
  const int ncol   = lane & 15;

  for (int kh = 0; kh < 3; ++kh) {
    const int hh = h + kh - 1;
    const bool hvalid = (hh >= 0) && (hh < H_DIM);

    for (int c0 = 0; c0 < C_IN; c0 += 32) {
      // ---- stage A: x row * style -> bf16 pairs, transposed into [wcol][chan] ----
      // 64 w x 16 channel-pairs = 1024 pairs; 16 per thread; one b32 LDS store each
      for (int e = tid; e < 64 * 16; e += 64) {
        int wc = e & 63;                 // consecutive lanes -> coalesced x reads
        int cp = e >> 6;                 // channel pair index
        int c  = c0 + 2 * cp;
        unsigned int pair = 0;
        if (hvalid) {
          const float* xr = &x[(((size_t)b * C_IN + c) * H_DIM + hh) * W_DIM + wc];
          float v0 = xr[0]                        * s[b * C_IN + c];
          float v1 = xr[(size_t)H_DIM * W_DIM]    * s[b * C_IN + c + 1];
          pair = (unsigned int)f2bf(v0) | ((unsigned int)f2bf(v1) << 16);
        }
        *(unsigned int*)&Ax[wc + 1][2 * cp] = pair;
      }
      if (tid < 32) { Ax[0][tid] = 0; Ax[65][tid] = 0; }   // horizontal SAME pad

      // ---- stage B: pre-converted bf16 weights, straight b128 copies ----
      // 3 kw x 64 o x 4 pieces = 768 uint4; 12 per thread
      for (int e = tid; e < 3 * 64 * 4; e += 64) {
        int piece = e & 3;
        int row   = e >> 2;              // kw*64 + o
        int kw    = row >> 6;
        int o     = row & 63;
        uint4 v = *(const uint4*)&wbf[(((size_t)(kh * 3 + kw) * C_OUT) + (o0 + o)) * C_IN
                                      + c0 + piece * 8];
        *(uint4*)&Bw[kw][o][piece * 8] = v;
      }

      // prefetch next channel chunk of x (global_prefetch_b8)
      if (hvalid && (c0 + 32) < C_IN) {
        __builtin_prefetch(
            &x[(((size_t)b * C_IN + (c0 + 32)) * H_DIM + hh) * W_DIM], 0, 0);
      }

      __syncthreads();

      // ---- 3 taps x (2 M-tiles x 4 N-tiles) = 24 WMMAs per chunk per wave ----
      #pragma unroll
      for (int kw = 0; kw < 3; ++kw) {
        const unsigned short* ar0 = &Ax[m0 + mlo + kw][0];
        const unsigned short* ar1 = &Ax[m0 + 16 + mlo + kw][0];
        v16bf a0 = frag_from(ar0 + khalfA, ar0 + khalfA + 16);
        v16bf a1 = frag_from(ar1 + khalfA, ar1 + khalfA + 16);

        #pragma unroll
        for (int nt = 0; nt < 4; ++nt) {
          const unsigned short* br = &Bw[kw][nt * 16 + ncol][khalfB];
          v16bf bf = frag_from(br, br + 16);
          acc[0][nt] = __builtin_amdgcn_wmma_f32_16x16x32_bf16(
              false, a0, false, bf, (short)0, acc[0][nt], false, false);
          acc[1][nt] = __builtin_amdgcn_wmma_f32_16x16x32_bf16(
              false, a1, false, bf, (short)0, acc[1][nt], false, false);
        }
      }
      __syncthreads();   // before LDS is overwritten
    }
  }

  // ---- epilogue: apply demod d[b,o], store per C-matrix VGPR layout ----
  const float* dv = dvec + b * C_OUT;
  const int mhalf = (lane & 16) ? 8 : 0;
  #pragma unroll
  for (int mt = 0; mt < 2; ++mt) {
    #pragma unroll
    for (int nt = 0; nt < 4; ++nt) {
      int o = o0 + nt * 16 + ncol;
      float dval = dv[o];
      float* orow = &out[(((size_t)b * C_OUT + o) * H_DIM + h) * W_DIM];
      #pragma unroll
      for (int r = 0; r < 8; ++r) {
        int wpos = m0 + mt * 16 + mhalf + r;
        orow[wpos] = acc[mt][nt][r] * dval;
      }
    }
  }
}

// ---------------- launcher ----------------
extern "C" void kernel_launch(void* const* d_in, const int* in_sizes, int n_in,
                              void* d_out, int out_size, void* d_ws, size_t ws_size,
                              hipStream_t stream) {
  const float* x    = (const float*)d_in[0];
  const float* y    = (const float*)d_in[1];
  const float* w    = (const float*)d_in[2];
  const float* fc_w = (const float*)d_in[3];
  const float* bias = (const float*)d_in[4];
  float* out = (float*)d_out;

  float* s   = (float*)d_ws;            // [B_N,  C_IN ]  = 8192 f32
  float* dv  = s  + B_N * C_IN;         // [B_N,  C_OUT]  = 8192 f32
  float* wsq = dv + B_N * C_OUT;        // [C_IN, C_OUT]  = 262144 f32
  unsigned short* wbf = (unsigned short*)(wsq + C_IN * C_OUT);  // [9][C_OUT][C_IN] bf16

  style_fc_kernel<<<(B_N * C_IN) / 256, 256, 0, stream>>>(y, fc_w, bias, s);
  wsq_kernel<<<(C_IN * C_OUT) / 256, 256, 0, stream>>>(w, wsq);
  demod_kernel<<<(B_N * C_OUT) / 256, 256, 0, stream>>>(s, wsq, dv);
  wprep_kernel<<<dim3(9, 2), 256, 0, stream>>>(w, wbf);

  dim3 grid(C_OUT / 64, H_DIM, B_N);
  modconv_wmma_kernel<<<grid, 64, 0, stream>>>(x, wbf, s, dv, out);
}
